// gcn_seq_84765474554102
// MI455X (gfx1250) — matrix operand order repaired and checked by reading, not compiled, not run on hardware
//
#include <hip/hip_runtime.h>

// ---------------- problem constants (match reference) ----------------
static constexpr int N   = 50000;
static constexpr int E   = 1600000;
static constexpr int G   = 1024;
static constexpr int C   = 128;     // IN = INS = OUT = 128
static constexpr int K   = 256;     // fan_in
static constexpr int L   = 5;
static constexpr float BN_EPS = 1e-5f;

typedef _Float16 v8h  __attribute__((ext_vector_type(8)));
typedef _Float16 v16h __attribute__((ext_vector_type(16)));
typedef float    v8f  __attribute__((ext_vector_type(8)));

// ---------------- utility kernels ----------------
__global__ __launch_bounds__(256) void k_zero(float* __restrict__ p, int n) {
    int i = blockIdx.x * 256 + threadIdx.x;
    if (i < n) p[i] = 0.0f;
}

__global__ __launch_bounds__(256) void k_fill1(float* __restrict__ p, int n) {
    int i = blockIdx.x * 256 + threadIdx.x;
    if (i < n) p[i] = 1.0f;
}

__global__ __launch_bounds__(256) void k_degcnt(const int* __restrict__ dst,
                                                float* __restrict__ deg, int e) {
    int i = blockIdx.x * 256 + threadIdx.x;
    if (i < e) atomicAdd(&deg[dst[i]], 1.0f);
}

__global__ __launch_bounds__(256) void k_rsqrt_inplace(float* __restrict__ p, int n) {
    int i = blockIdx.x * 256 + threadIdx.x;
    if (i < n) p[i] = rsqrtf(p[i]);
}

// Convert conv_w [L][K][C] (f32, row-major) -> Wt [L][C][K] (f16, transposed)
__global__ __launch_bounds__(256) void k_wconv(const float* __restrict__ W,
                                               _Float16* __restrict__ Wt) {
    int idx = blockIdx.x * 256 + threadIdx.x;          // < L*K*C
    int l   = idx / (K * C);
    int rem = idx % (K * C);
    int k   = rem / C;
    int n   = rem % C;
    Wt[(size_t)l * C * K + (size_t)n * K + k] = (_Float16)W[idx];
}

// ---------------- fused concat + f16-convert + WMMA GEMM ----------------
// Block: 256 threads = 8 waves. Block handles 16 node-rows x all 128 out cols.
// Wave w computes the 16x16 tile at columns [16w, 16w+16).
__global__ __launch_bounds__(256) void k_gemm(const float* __restrict__ Hin,
                                              const float* __restrict__ insL,
                                              const int*   __restrict__ batch,
                                              const _Float16* __restrict__ Wt,
                                              float* __restrict__ T) {
    __shared__ _Float16 Ah[16][K + 8];   // 16 rows x 256 (+8 pad) f16
    const int tid  = threadIdx.x;
    const int tile = blockIdx.x;         // 0..N/16-1

    // ---- stage xc = [h | ins[batch]] tile into LDS as f16 ----
    {
        const int r   = tid >> 4;        // 0..15 row within tile
        const int cg  = tid & 15;        // 16 col-groups of 16
        const int row = tile * 16 + r;
        const int b   = batch[row];
        const float* hrow = Hin  + (size_t)row * C;
        const float* irow = insL + (size_t)b   * C;
#pragma unroll
        for (int u = 0; u < 16; ++u) {
            const int col = cg * 16 + u;
            const float v = (col < C) ? hrow[col] : irow[col - C];
            Ah[r][col] = (_Float16)v;
        }
    }
    __syncthreads();

    const int lane = tid & 31;
    const int wave = tid >> 5;           // 0..7 -> column tile
    const int m    = lane & 15;
    const int hi   = lane >> 4;          // 0: lanes 0-15, 1: lanes 16-31
    const int n    = wave * 16 + m;      // output column owned by this lane (B/C/D)

    v8f acc = {};
    const _Float16* wrow = Wt + (size_t)n * K;   // W^T row: contiguous in K

    #pragma unroll
    for (int kc = 0; kc < K / 32; ++kc) {
        // A fragment (16-bit A 16x32 layout, ISA 7.12.2): lane half selects kb
        const int kb = hi ? 8 : 0;
        const v8h* ap = (const v8h*)&Ah[m][kc * 32 + kb];
        v8h a0 = ap[0];          // K = kb .. kb+7
        v8h a1 = ap[2];          // K = 16+kb .. 16+kb+7   (+16 halves)
        v16h a = __builtin_shufflevector(a0, a1,
                 0, 1, 2, 3, 4, 5, 6, 7, 8, 9, 10, 11, 12, 13, 14, 15);
        // B fragment: same K order as A (koff..koff+7, 16+koff..16+koff+7)
        const int koff = hi ? 8 : 0;
        const v8h* bp = (const v8h*)(wrow + kc * 32 + koff);
        v8h b0 = bp[0];
        v8h b1 = bp[2];
        v16h bf = __builtin_shufflevector(b0, b1,
                 0, 1, 2, 3, 4, 5, 6, 7, 8, 9, 10, 11, 12, 13, 14, 15);
        acc = __builtin_amdgcn_wmma_f32_16x16x32_f16(
                  false, a, false, bf, (short)0, acc, false, false);
    }

    // C/D layout: VGPR r -> row M = r + 8*hi, col n = lane&15 (+tile offset)
#pragma unroll
    for (int r = 0; r < 8; ++r) {
        const int M = r + (hi ? 8 : 0);
        T[(size_t)(tile * 16 + M) * C + n] = acc[r];
    }
}

// ---------------- edge scatter: agg[dst] += t[src] * dinv[src]*dinv[dst] ----
// One wave per edge; each lane handles 4 contiguous channels (float4).
__global__ __launch_bounds__(256) void k_scatter(const float* __restrict__ T,
                                                 const int* __restrict__ src,
                                                 const int* __restrict__ dst,
                                                 const float* __restrict__ dinv,
                                                 float* __restrict__ agg) {
    const int tid  = threadIdx.x;
    const int lane = tid & 31;
    const int e    = blockIdx.x * 8 + (tid >> 5);
    const int s = src[e];
    const int d = dst[e];
    const float c = dinv[s] * dinv[d];
    const float4 v = *(const float4*)(T + (size_t)s * C + lane * 4);
    float* o = agg + (size_t)d * C + lane * 4;
    atomicAdd(o + 0, v.x * c);
    atomicAdd(o + 1, v.y * c);
    atomicAdd(o + 2, v.z * c);
    atomicAdd(o + 3, v.w * c);
}

// ---------------- combine: h = agg + t*dinv^2 + b; accumulate BN stats ------
// 128 threads (one per channel); each block handles 50 rows -> 1000 blocks.
__global__ __launch_bounds__(128) void k_combine(const float* __restrict__ T,
                                                 const float* __restrict__ agg,
                                                 const float* __restrict__ dinv,
                                                 const float* __restrict__ bias,
                                                 float* __restrict__ Hout,
                                                 float* __restrict__ stats,
                                                 int doStats) {
    const int c  = threadIdx.x;
    const int r0 = blockIdx.x * 50;
    const float b = bias[c];
    float s = 0.0f, sq = 0.0f;
    for (int r = r0; r < r0 + 50; ++r) {
        const float di = dinv[r];
        const float v = agg[(size_t)r * C + c] + T[(size_t)r * C + c] * (di * di) + b;
        Hout[(size_t)r * C + c] = v;
        s  += v;
        sq += v * v;
    }
    if (doStats) {
        atomicAdd(&stats[c], s);
        atomicAdd(&stats[C + c], sq);
    }
}

// ---------------- BN (batch stats) + ReLU, in place ----------------
__global__ __launch_bounds__(256) void k_bn_relu(float* __restrict__ H,
                                                 const float* __restrict__ stats,
                                                 const float* __restrict__ gamma,
                                                 const float* __restrict__ beta) {
    const int idx = blockIdx.x * 256 + threadIdx.x;   // < N*C (exact)
    const int c = idx & (C - 1);
    const float inv_n = 1.0f / (float)N;
    const float mean = stats[c] * inv_n;
    const float var  = stats[C + c] * inv_n - mean * mean;
    const float sc   = gamma[c] * rsqrtf(var + BN_EPS);
    const float v    = sc * (H[idx] - mean) + beta[c];
    H[idx] = v > 0.0f ? v : 0.0f;
}

// ---------------- driver ----------------
extern "C" void kernel_launch(void* const* d_in, const int* in_sizes, int n_in,
                              void* d_out, int out_size, void* d_ws, size_t ws_size,
                              hipStream_t stream) {
    const float* x     = (const float*)d_in[0];
    const int*   ei    = (const int*)d_in[1];
    const float* instr = (const float*)d_in[2];
    const int*   batch = (const int*)d_in[3];
    const float* convw = (const float*)d_in[4];
    const float* convb = (const float*)d_in[5];
    const float* gamma = (const float*)d_in[6];
    const float* beta  = (const float*)d_in[7];
    float*       out   = (float*)d_out;

    const int* esrc = ei;
    const int* edst = ei + E;

    // workspace layout
    float* t     = (float*)d_ws;                 // N*C
    float* agg   = t    + (size_t)N * C;         // N*C
    float* hbuf  = agg  + (size_t)N * C;         // N*C
    float* dinv  = hbuf + (size_t)N * C;         // N
    float* stats = dinv + N;                     // 2*C
    _Float16* Wt = (_Float16*)(stats + 2 * C);   // L*C*K halves

    // one-time prep: weights -> f16 transposed; degree -> dinv
    k_wconv<<<(L * K * C) / 256, 256, 0, stream>>>(convw, Wt);
    k_fill1<<<(N + 255) / 256, 256, 0, stream>>>(dinv, N);
    k_degcnt<<<E / 256, 256, 0, stream>>>(edst, dinv, E);
    k_rsqrt_inplace<<<(N + 255) / 256, 256, 0, stream>>>(dinv, N);

    for (int i = 0; i < L; ++i) {
        const float* hin  = (i == 0) ? x : hbuf;
        const bool   last = (i == L - 1);

        k_zero<<<(N * C) / 256, 256, 0, stream>>>(agg, N * C);
        k_zero<<<1, 256, 0, stream>>>(stats, 2 * C);

        k_gemm<<<N / 16, 256, 0, stream>>>(hin, instr + (size_t)i * G * C, batch,
                                           Wt + (size_t)i * C * K, t);
        k_scatter<<<E / 8, 256, 0, stream>>>(t, esrc, edst, dinv, agg);
        k_combine<<<N / 50, 128, 0, stream>>>(t, agg, dinv, convb + i * C,
                                              last ? out : hbuf, stats, last ? 0 : 1);
        if (!last)
            k_bn_relu<<<(N * C) / 256, 256, 0, stream>>>(hbuf, stats,
                                                         gamma + i * C, beta + i * C);
    }
}
